// InterSlotCommunication_21242908246209
// MI455X (gfx1250) — matrix operand order repaired and checked by reading, not compiled, not run on hardware
//
#include <hip/hip_runtime.h>

// ---------------------------------------------------------------------------
// Fused multihead self-attention over 3 slots + residual + LayerNorm.
//   B=32768, S=3, H=1024, heads=16, dh=64.
// Compute-bound (~0.83 TFLOP): big GEMMs run on v_wmma_f32_16x16x32_bf16.
// ---------------------------------------------------------------------------

typedef __attribute__((ext_vector_type(16))) __bf16 v16bf;
typedef __attribute__((ext_vector_type(8)))  __bf16 v8bf;
typedef __attribute__((ext_vector_type(8)))  float  v8f;

#define HID        1024
#define ROWS       48          // 16 batches * 3 slots per workgroup
#define SA_STRIDE  1032        // bf16 elements, padded to dodge LDS bank conflicts
#define QS         384         // qkvh scratch row stride (2 heads * 192 floats)

// LDS layout (bytes). Phase1 region (sA+qkvh+scores) is aliased by fp32 y in
// phase2/3; ctx persists across the phase boundary.
#define OFF_SA     0                         // bf16 [48][1032]  -> 99072 B
#define OFF_QKVH   99072                     // f32  [48][384]   -> 73728 B
#define OFF_SCORES 172800                    // f32  [16][2][9]  ->  1152 B
#define OFF_Y      0                         // f32  [48][1024]  -> 196608 B (aliases above)
#define OFF_CTX    196608                    // bf16 [48][1032]  -> 99072 B
#define TOTAL_LDS  295680                    // 288.75 KB <= 320 KB/WGP

static __device__ __forceinline__ unsigned short f2bf(float f) {
  unsigned u = __float_as_uint(f);
  u += 0x7FFFu + ((u >> 16) & 1u);           // round-to-nearest-even
  return (unsigned short)(u >> 16);
}

static __device__ __forceinline__ v16bf ldfrag(const __bf16* p0, const __bf16* p1) {
  v16bf r;
  ((v8bf*)&r)[0] = *(const v8bf*)p0;
  ((v8bf*)&r)[1] = *(const v8bf*)p1;
  return r;
}

// fp32 -> bf16 weight pre-conversion (weights then live in L2 for the run)
__global__ void cvt_bf16_kernel(const float* __restrict__ src,
                                unsigned short* __restrict__ dst, int n4) {
  int i = blockIdx.x * blockDim.x + threadIdx.x;
  if (i >= n4) return;
  float4 f = ((const float4*)src)[i];
  uint2 u;
  u.x = (unsigned)f2bf(f.x) | ((unsigned)f2bf(f.y) << 16);
  u.y = (unsigned)f2bf(f.z) | ((unsigned)f2bf(f.w) << 16);
  ((uint2*)dst)[i] = u;
}

__global__ __launch_bounds__(256, 1)
void fused_attn_ln_kernel(const float* __restrict__ slots,     // [B,3,1024] f32
                          const __bf16* __restrict__ Wqkv,     // [3072][1024] bf16
                          const __bf16* __restrict__ Wout,     // [1024][1024] bf16
                          const float* __restrict__ in_b,      // [3072]
                          const float* __restrict__ outb,      // [1024]
                          const float* __restrict__ ln_g,      // [1024]
                          const float* __restrict__ ln_b,      // [1024]
                          float* __restrict__ out)             // [B,3,1024] f32
{
  extern __shared__ char smem[];
  unsigned short* sA_us  = (unsigned short*)(smem + OFF_SA);
  const __bf16*   sAb    = (const __bf16*)(smem + OFF_SA);
  float*          qkvh   = (float*)(smem + OFF_QKVH);
  float*          scoresL= (float*)(smem + OFF_SCORES);
  float*          yL     = (float*)(smem + OFF_Y);
  unsigned short* ctx_us = (unsigned short*)(smem + OFF_CTX);
  const __bf16*   ctxb   = (const __bf16*)(smem + OFF_CTX);

  const int tid  = threadIdx.x;
  const int wv   = tid >> 5;
  const int lane = tid & 31;
  const int lr   = lane & 15;          // row(M)/col(N) index within fragment
  const int lh   = (lane >> 4) & 1;    // half select per ISA 16-bit layouts
  const size_t rowBaseG = (size_t)blockIdx.x * ROWS;   // global M row base
  const v8f vz = {0.f,0.f,0.f,0.f,0.f,0.f,0.f,0.f};

  // ---- Phase 0: stage slots tile [48][1024] as bf16 into LDS -------------
  {
    const float4* src = (const float4*)(slots + rowBaseG * HID);
#pragma unroll 4
    for (int i = 0; i < 48; ++i) {
      int j = tid + 256 * i;                 // 12288 float4s total
      float4 f = src[j];
      int e = j * 4, row = e >> 10, col = e & 1023;
      uint2 u;
      u.x = (unsigned)f2bf(f.x) | ((unsigned)f2bf(f.y) << 16);
      u.y = (unsigned)f2bf(f.z) | ((unsigned)f2bf(f.w) << 16);
      *(uint2*)(sA_us + row * SA_STRIDE + col) = u;
    }
  }
  __syncthreads();

  // ---- Phase 1: per head-pair QKV GEMM (WMMA) + tiny attention -----------
  for (int hp = 0; hp < 8; ++hp) {
    // 24 N-panels of 16 (2 heads * {q,k,v} * 4), 3 per wave; each panel
    // carries 3 M-tiles so every B fragment is reused 3x.
#pragma unroll 1
    for (int j = 0; j < 3; ++j) {
      const int p     = wv + 8 * j;
      const int hh    = p / 12;
      const int nt    = p % 12;
      const int which = nt >> 2;          // 0=q 1=k 2=v
      const int nn    = nt & 3;
      const int eBase = which * HID + (hp * 2 + hh) * 64 + nn * 16;
      const int cBase = hh * 192 + which * 64 + nn * 16;

      v8f acc0 = vz, acc1 = vz, acc2 = vz;
      const __bf16* Wp = Wqkv + (size_t)(eBase + lr) * HID + lh * 8;
      const __bf16* Ap = sAb + lr * SA_STRIDE + lh * 8;
#pragma unroll 4
      for (int k = 0; k < HID; k += 32) {
        v16bf b  = ldfrag(Wp + k, Wp + k + 16);
        v16bf a0 = ldfrag(Ap + k,                  Ap + k + 16);
        v16bf a1 = ldfrag(Ap + 16 * SA_STRIDE + k, Ap + 16 * SA_STRIDE + k + 16);
        v16bf a2 = ldfrag(Ap + 32 * SA_STRIDE + k, Ap + 32 * SA_STRIDE + k + 16);
        acc0 = __builtin_amdgcn_wmma_f32_16x16x32_bf16(false, a0, false, b, (short)0, acc0, false, false);
        acc1 = __builtin_amdgcn_wmma_f32_16x16x32_bf16(false, a1, false, b, (short)0, acc1, false, false);
        acc2 = __builtin_amdgcn_wmma_f32_16x16x32_bf16(false, a2, false, b, (short)0, acc2, false, false);
      }
      const float bias = in_b[eBase + lr];
      const float* a0f = (const float*)&acc0;
      const float* a1f = (const float*)&acc1;
      const float* a2f = (const float*)&acc2;
#pragma unroll
      for (int r = 0; r < 8; ++r) {
        int row = r + lh * 8;
        qkvh[(0  + row) * QS + cBase + lr] = a0f[r] + bias;
        qkvh[(16 + row) * QS + cBase + lr] = a1f[r] + bias;
        qkvh[(32 + row) * QS + cBase + lr] = a2f[r] + bias;
      }
    }
    __syncthreads();

    // scores: 16 batches * 2 heads * 9 pairs, dot64 each
    for (int idx = tid; idx < 288; idx += 256) {
      int bb = idx / 18, rem = idx % 18, hh = rem / 9, pair = rem % 9;
      int sq = pair / 3, sk = pair % 3;
      const float* qp = qkvh + (bb * 3 + sq) * QS + hh * 192;
      const float* kp = qkvh + (bb * 3 + sk) * QS + hh * 192 + 64;
      float s = 0.f;
#pragma unroll 8
      for (int d = 0; d < 64; ++d) s += qp[d] * kp[d];
      scoresL[(bb * 2 + hh) * 9 + pair] = s * 0.125f;   // 1/sqrt(64)
    }
    __syncthreads();

    // softmax over k (3 values) for 48 rows * 2 heads
    if (tid < 96) {
      int bb = tid / 6, r6 = tid % 6, hh = r6 / 3, sq = r6 % 3;
      float* sp = scoresL + (bb * 2 + hh) * 9 + sq * 3;
      float m = fmaxf(sp[0], fmaxf(sp[1], sp[2]));
      float e0 = __expf(sp[0] - m), e1 = __expf(sp[1] - m), e2 = __expf(sp[2] - m);
      float inv = 1.f / (e0 + e1 + e2);
      sp[0] = e0 * inv; sp[1] = e1 * inv; sp[2] = e2 * inv;
    }
    __syncthreads();

    // ctx = attn @ v  -> bf16 ctx tile columns for these 2 heads
    for (int o = tid; o < 6144; o += 256) {
      int d = o & 63, r96 = o >> 6;
      int bb = r96 / 6, r6 = r96 % 6, hh = r6 / 3, sq = r6 % 3;
      const float* ap = scoresL + (bb * 2 + hh) * 9 + sq * 3;
      const float* vp = qkvh + (bb * 3) * QS + hh * 192 + 128 + d;
      float s = ap[0] * vp[0] + ap[1] * vp[QS] + ap[2] * vp[2 * QS];
      ctx_us[(bb * 3 + sq) * SA_STRIDE + (hp * 2 + hh) * 64 + d] = f2bf(s);
    }
    __syncthreads();   // protects qkvh reuse next iter / ctx before phase 2
  }

  // ---- Phase 2: out-proj GEMM (WMMA) + bias + residual into fp32 y -------
#pragma unroll 1
  for (int nt = wv; nt < 64; nt += 8) {
    const int nBase = nt * 16;
    v8f acc0 = vz, acc1 = vz, acc2 = vz;
    const __bf16* Wp = Wout + (size_t)(nBase + lr) * HID + lh * 8;
    const __bf16* Ap = ctxb + lr * SA_STRIDE + lh * 8;
#pragma unroll 4
    for (int k = 0; k < HID; k += 32) {
      v16bf b  = ldfrag(Wp + k, Wp + k + 16);
      v16bf a0 = ldfrag(Ap + k,                  Ap + k + 16);
      v16bf a1 = ldfrag(Ap + 16 * SA_STRIDE + k, Ap + 16 * SA_STRIDE + k + 16);
      v16bf a2 = ldfrag(Ap + 32 * SA_STRIDE + k, Ap + 32 * SA_STRIDE + k + 16);
      acc0 = __builtin_amdgcn_wmma_f32_16x16x32_bf16(false, a0, false, b, (short)0, acc0, false, false);
      acc1 = __builtin_amdgcn_wmma_f32_16x16x32_bf16(false, a1, false, b, (short)0, acc1, false, false);
      acc2 = __builtin_amdgcn_wmma_f32_16x16x32_bf16(false, a2, false, b, (short)0, acc2, false, false);
    }
    const int   n    = nBase + lr;
    const float bias = outb[n];
    const float* af[3] = {(const float*)&acc0, (const float*)&acc1, (const float*)&acc2};
#pragma unroll
    for (int mt = 0; mt < 3; ++mt) {
#pragma unroll
      for (int r = 0; r < 8; ++r) {
        int row = mt * 16 + r + lh * 8;
        float resid = slots[(rowBaseG + row) * HID + n];
        yL[row * HID + n] = af[mt][r] + bias + resid;
      }
    }
  }
  __syncthreads();

  // ---- Phase 3: LayerNorm per row (wave32 shuffle reduction) -------------
  for (int rr = wv; rr < ROWS; rr += 8) {
    float vals[32];
    float s = 0.f, s2 = 0.f;
#pragma unroll
    for (int i = 0; i < 32; ++i) {
      float v = yL[rr * HID + i * 32 + lane];
      vals[i] = v;
      s += v; s2 += v * v;
    }
#pragma unroll
    for (int off = 16; off > 0; off >>= 1) {
      s  += __shfl_xor(s,  off);
      s2 += __shfl_xor(s2, off);
    }
    const float mean = s * (1.f / 1024.f);
    const float var  = s2 * (1.f / 1024.f) - mean * mean;
    const float rstd = rsqrtf(var + 1e-5f);
    float* op = out + (rowBaseG + rr) * HID;
#pragma unroll
    for (int i = 0; i < 32; ++i) {
      int n = i * 32 + lane;
      op[n] = (vals[i] - mean) * rstd * ln_g[n] + ln_b[n];
    }
  }
}

extern "C" void kernel_launch(void* const* d_in, const int* in_sizes, int n_in,
                              void* d_out, int out_size, void* d_ws, size_t ws_size,
                              hipStream_t stream) {
  const float* slots = (const float*)d_in[0];   // [32768,3,1024]
  const float* ipw   = (const float*)d_in[1];   // [3072,1024]
  const float* ipb   = (const float*)d_in[2];   // [3072]
  const float* outw  = (const float*)d_in[3];   // [1024,1024]
  const float* outb  = (const float*)d_in[4];   // [1024]
  const float* ln_g  = (const float*)d_in[5];   // [1024]
  const float* ln_b  = (const float*)d_in[6];   // [1024]
  float* out = (float*)d_out;

  // ws: bf16 in_proj_w (3,145,728 elems) then bf16 out_w (1,048,576 elems)
  unsigned short* wq = (unsigned short*)d_ws;
  unsigned short* wo = wq + 3145728;
  cvt_bf16_kernel<<<3072, 256, 0, stream>>>(ipw,  wq, 786432);
  cvt_bf16_kernel<<<1024, 256, 0, stream>>>(outw, wo, 262144);

  hipFuncSetAttribute((const void*)fused_attn_ln_kernel,
                      hipFuncAttributeMaxDynamicSharedMemorySize, TOTAL_LDS);
  fused_attn_ln_kernel<<<2048, 256, TOTAL_LDS, stream>>>(
      slots, (const __bf16*)wq, (const __bf16*)wo, ipb, outb, ln_g, ln_b, out);
}
